// CausalSelfAttentionHead_7928509628740
// MI455X (gfx1250) — matrix operand order repaired and checked by reading, not compiled, not run on hardware
//
#include <hip/hip_runtime.h>

// ---------------------------------------------------------------------------
// Causal self-attention head with softmax over the QUERY axis (axis=-2).
//   k = x@Wk, q = x@Wq, v = x@Wv             (bf16 WMMA, f32 accum)
//   S = q k^T * 1/8, causal mask (q >= k)
//   P[:,k] = exp(S[:,k]) / sum_{q>=k} exp(S[q,k])   (per-COLUMN normalization)
//   out = P @ v
// Per-key-column normalizer is precomputed (no flash-style online rescale).
// All GEMMs use v_wmma_f32_16x16x32_bf16.  bf16 converts use native HW cvt.
// ---------------------------------------------------------------------------

#define TB  4096   // T
#define CE  1024   // C_EMB
#define CHD 64     // C_HEAD
#define NB  4      // B

typedef unsigned short bf16raw;
typedef __attribute__((ext_vector_type(16))) __bf16 bf16x16;
typedef __attribute__((ext_vector_type(8)))  __bf16 bf16x8;
typedef __attribute__((ext_vector_type(8)))  float  f32x8;

// native hardware f32 -> bf16 conversion (RNE on gfx1250)
__device__ inline __bf16 f2bf(float f) { return (__bf16)f; }
__device__ inline bf16raw f2bfr(float f) {
  __bf16 h = (__bf16)f;
  return __builtin_bit_cast(bf16raw, h);
}
// load 16 bf16 that are contiguous in memory
__device__ inline bf16x16 ld16(const __bf16* p) {
  return *reinterpret_cast<const bf16x16*>(p);
}
// A-fragment style load: 8 bf16 at p, 8 bf16 at p+16 (K halves split by 16)
__device__ inline bf16x16 ld8x2(const __bf16* p) {
  bf16x8 lo = *reinterpret_cast<const bf16x8*>(p);
  bf16x8 hi = *reinterpret_cast<const bf16x8*>(p + 16);
  return __builtin_shufflevector(lo, hi, 0,1,2,3,4,5,6,7,8,9,10,11,12,13,14,15);
}
__device__ inline f32x8 wmma_bf16(bf16x16 a, bf16x16 b, f32x8 c) {
  return __builtin_amdgcn_wmma_f32_16x16x32_bf16(false, a, false, b,
                                                 (short)0, c, false, false);
}

// ---------------------------------------------------------------------------
// Kernel 0: zero the denominator accumulator
// ---------------------------------------------------------------------------
__global__ void k_zero(float* p, int n) {
  int i = blockIdx.x * blockDim.x + threadIdx.x;
  if (i < n) p[i] = 0.f;
}

// ---------------------------------------------------------------------------
// Kernel 1: projections.  One wave computes a 16-row tile of Q, K, V (16x64
// each).  W slabs (32x64 x 3 mats, bf16) staged through LDS each K-step.
// Q,K stored row-major bf16 [B*T][64]; V stored TRANSPOSED bf16 [B][64][T]
// so that the PV WMMA B-fragments are two contiguous 16B loads later.
// ---------------------------------------------------------------------------
__global__ __launch_bounds__(256) void k_proj(
    const float* __restrict__ x,
    const float* __restrict__ Wq, const float* __restrict__ Wk,
    const float* __restrict__ Wv,
    bf16raw* __restrict__ Qb, bf16raw* __restrict__ Kb,
    bf16raw* __restrict__ Vt)
{
  __shared__ __bf16 wslab[3 * 64 * 32];          // [mat][n(64)][k(32)] bf16
  const int tid  = threadIdx.x;
  const int wv   = tid >> 5;
  const int lane = tid & 31;
  const int m16  = lane & 15;
  const int h    = lane >> 4;
  const int rowBase = blockIdx.x * 128 + wv * 16;

  const float* Wmat[3] = { Wq, Wk, Wv };

  f32x8 z8 = {0.f,0.f,0.f,0.f,0.f,0.f,0.f,0.f};
  f32x8 acc[12];
  #pragma unroll
  for (int i = 0; i < 12; ++i) acc[i] = z8;

  for (int kk = 0; kk < CE; kk += 32) {
    __syncthreads();                              // protect previous reads
    // cooperative stage: LDS[m][n][k] = bf16(W_m[kk+k][n])
    for (int idx = tid; idx < 3 * 32 * 64; idx += 256) {
      int n = idx & 63;
      int k = (idx >> 6) & 31;
      int m = idx >> 11;
      wslab[m * 2048 + n * 32 + k] = f2bf(Wmat[m][(size_t)(kk + k) * CHD + n]);
    }
    __syncthreads();

    // A fragment of x (fp32 -> bf16): row = rowBase+m16, K = h*8+{0..7,16..23}
    const float* ax = x + (size_t)(rowBase + m16) * CE + kk + h * 8;
    bf16x16 a;
    #pragma unroll
    for (int j = 0; j < 8; ++j) a[j] = f2bf(ax[j]);
    #pragma unroll
    for (int j = 0; j < 8; ++j) a[8 + j] = f2bf(ax[16 + j]);

    #pragma unroll
    for (int m = 0; m < 3; ++m) {
      #pragma unroll
      for (int t = 0; t < 4; ++t) {
        // B frag: lane holds col n = t*16+m16, K = h*8+{0..7,16..23}
        const __bf16* bp = &wslab[m * 2048 + (t * 16 + m16) * 32 + h * 8];
        bf16x16 bfrag = ld8x2(bp);
        acc[m * 4 + t] = wmma_bf16(a, bfrag, acc[m * 4 + t]);
      }
    }
  }

  // store: C layout -> lane holds N=m16, rows M=r+8*h
  #pragma unroll
  for (int m = 0; m < 3; ++m) {
    #pragma unroll
    for (int t = 0; t < 4; ++t) {
      #pragma unroll
      for (int r = 0; r < 8; ++r) {
        int M = r + 8 * h;
        int N = t * 16 + m16;
        int grow = rowBase + M;                   // global row in [0, B*T)
        bf16raw hv = f2bfr(acc[m * 4 + t][r]);
        if (m == 0)      Qb[(size_t)grow * CHD + N] = hv;
        else if (m == 1) Kb[(size_t)grow * CHD + N] = hv;
        else {
          int b  = grow >> 12;                    // /T
          int tr = grow & (TB - 1);
          Vt[((size_t)(b * CHD + N)) * TB + tr] = hv;
        }
      }
    }
  }
}

// ---------------------------------------------------------------------------
// Kernel 2: per-key-column denominators  denom[b,k] = sum_{q>=k} exp(s/8).
// One wave owns (b, key-tile kt, q-chunk ch); q range split into 8 chunks to
// balance the triangular schedule; partial column sums merged via
// global atomic f32 add.
// ---------------------------------------------------------------------------
__global__ __launch_bounds__(256) void k_denom(
    const bf16raw* __restrict__ Qb, const bf16raw* __restrict__ Kb,
    float* __restrict__ denom)
{
  const int tid  = threadIdx.x;
  const int wv   = tid >> 5;
  const int lane = tid & 31;
  const int m16  = lane & 15;
  const int h    = lane >> 4;

  int w   = blockIdx.x * 8 + wv;                  // 8192 waves total
  int b   = w >> 11;                              // / (256*8)
  int rem = w & 2047;
  int kt  = rem >> 3;
  int ch  = rem & 7;
  int qlo = kt > ch * 32 ? kt : ch * 32;
  int qhi = (ch * 32 + 32) < 256 ? (ch * 32 + 32) : 256;

  // B fragments of K^T: B[c,k] = K[key][c];  lane: N=key=m16, K=c=h*8+{0..7,16..23}
  const __bf16* Kp =
      (const __bf16*)Kb + ((size_t)b * TB + kt * 16 + m16) * CHD + h * 8;
  bf16x16 bk0 = ld8x2(Kp);
  bf16x16 bk1 = ld8x2(Kp + 32);

  const __bf16* Qbase = (const __bf16*)Qb + (size_t)b * TB * CHD;
  const int gk = kt * 16 + m16;
  f32x8 z8 = {0.f,0.f,0.f,0.f,0.f,0.f,0.f,0.f};
  float csum = 0.f;

  for (int qt = qlo; qt < qhi; ++qt) {
    const __bf16* ap = Qbase + (size_t)(qt * 16 + m16) * CHD + h * 8;
    bf16x16 a0 = ld8x2(ap);                       // c = 0..31
    bf16x16 a1 = ld8x2(ap + 32);                  // c = 32..63
    f32x8 c = z8;
    c = wmma_bf16(a0, bk0, c);
    c = wmma_bf16(a1, bk1, c);
    #pragma unroll
    for (int r = 0; r < 8; ++r) {
      int gq = qt * 16 + r + 8 * h;
      if (gq >= gk) csum += __expf(c[r] * 0.125f);
    }
  }
  // lanes l and l+16 hold the two row-halves of the same column
  csum += __shfl_xor(csum, 16);
  if (lane < 16 && qlo < qhi)
    atomicAdd(&denom[b * TB + kt * 16 + lane], csum);
}

// ---------------------------------------------------------------------------
// Kernel 3: output.  One wave owns a 16-query tile; loops over 32-key blocks:
//   S = Q K^T (2 WMMAs per 16-key subtile), P = exp(S)*mask/denom[k],
//   P repacked C-layout -> A-layout through a 1KB per-wave LDS bounce,
//   O += P V (4 WMMAs, V consumed transposed -> contiguous B-frag loads).
// ---------------------------------------------------------------------------
__global__ __launch_bounds__(256) void k_out(
    const bf16raw* __restrict__ Qb, const bf16raw* __restrict__ Kb,
    const bf16raw* __restrict__ Vt, const float* __restrict__ denom,
    float* __restrict__ out)
{
  __shared__ __bf16 pbuf[8 * 512];                // [wave][16 rows][32 keys]
  const int tid  = threadIdx.x;
  const int wv   = tid >> 5;
  const int lane = tid & 31;
  const int m16  = lane & 15;
  const int h    = lane >> 4;

  int w  = blockIdx.x * 8 + wv;                   // 1024 waves total
  int b  = w >> 8;
  int qt = w & 255;

  __bf16* P = pbuf + wv * 512;
  const __bf16 zb = __builtin_bit_cast(__bf16, (unsigned short)0);

  // Q A-fragments, invariant across the key loop
  const __bf16* Qp =
      (const __bf16*)Qb + ((size_t)b * TB + qt * 16 + m16) * CHD + h * 8;
  bf16x16 aq0 = ld8x2(Qp);
  bf16x16 aq1 = ld8x2(Qp + 32);

  f32x8 z8 = {0.f,0.f,0.f,0.f,0.f,0.f,0.f,0.f};
  f32x8 accO[4];
  #pragma unroll
  for (int t = 0; t < 4; ++t) accO[t] = z8;

  const int qmax = qt * 16 + 15;

  for (int k0 = 0; k0 <= qmax; k0 += 32) {
    #pragma unroll
    for (int s = 0; s < 2; ++s) {
      int kb = k0 + s * 16;
      if (kb <= qmax) {
        const __bf16* Kp =
            (const __bf16*)Kb + ((size_t)b * TB + kb + m16) * CHD + h * 8;
        bf16x16 b0 = ld8x2(Kp);
        bf16x16 b1 = ld8x2(Kp + 32);
        f32x8 c = z8;
        c = wmma_bf16(aq0, b0, c);
        c = wmma_bf16(aq1, b1, c);
        int   gk = kb + m16;
        float rd = 1.0f / denom[b * TB + gk];
        #pragma unroll
        for (int r = 0; r < 8; ++r) {
          int gq = qt * 16 + r + 8 * h;
          float v = (gq >= gk) ? __expf(c[r] * 0.125f) * rd : 0.f;
          P[(r + 8 * h) * 32 + s * 16 + m16] = f2bf(v);   // ds_store_b16
        }
      } else {
        #pragma unroll
        for (int r = 0; r < 8; ++r)
          P[(r + 8 * h) * 32 + s * 16 + m16] = zb;
      }
    }
    // Repack P as WMMA A fragment: lane row = m16, K = h*8 + {0..7,16..23}
    const __bf16* pp = P + m16 * 32 + h * 8;
    bf16x16 ap = ld8x2(pp);                       // ds_load_b128 x2
    #pragma unroll
    for (int t = 0; t < 4; ++t) {
      // V^T layout: B[key,c] contiguous along key -> 2x16B loads, key = h*8+{0..7,16..23}
      const __bf16* vp =
          (const __bf16*)Vt + ((size_t)(b * CHD + t * 16 + m16)) * TB + k0 + h * 8;
      bf16x16 bv = ld8x2(vp);
      accO[t] = wmma_bf16(ap, bv, accO[t]);
    }
  }

  #pragma unroll
  for (int t = 0; t < 4; ++t) {
    #pragma unroll
    for (int r = 0; r < 8; ++r) {
      int gq = qt * 16 + r + 8 * h;
      out[((size_t)b * TB + gq) * CHD + t * 16 + m16] = accO[t][r];
    }
  }
}

// ---------------------------------------------------------------------------
// Launcher.  Workspace:
//   [0,2MB)     Qb  bf16 [B*T][64]
//   [2MB,4MB)   Kb  bf16 [B*T][64]
//   [4MB,6MB)   Vt  bf16 [B][64][T]   (transposed V)
//   [6MB,+64KB) denom f32 [B*T]
// ---------------------------------------------------------------------------
extern "C" void kernel_launch(void* const* d_in, const int* in_sizes, int n_in,
                              void* d_out, int out_size, void* d_ws,
                              size_t ws_size, hipStream_t stream) {
  const float* x  = (const float*)d_in[0];
  const float* Wk = (const float*)d_in[1];
  const float* Wq = (const float*)d_in[2];
  const float* Wv = (const float*)d_in[3];
  float* out = (float*)d_out;

  char* ws = (char*)d_ws;
  bf16raw* Qb    = (bf16raw*)(ws);
  bf16raw* Kb    = (bf16raw*)(ws + (size_t)2 * 1024 * 1024);
  bf16raw* Vt    = (bf16raw*)(ws + (size_t)4 * 1024 * 1024);
  float*   denom = (float*)  (ws + (size_t)6 * 1024 * 1024);

  // projections: 16384 rows / (16 rows/wave * 8 waves/block) = 128 blocks
  k_proj<<<128, 256, 0, stream>>>(x, Wq, Wk, Wv, Qb, Kb, Vt);

  // denom = 0, then accumulate column sums
  k_zero<<<(NB * TB + 255) / 256, 256, 0, stream>>>(denom, NB * TB);
  // 4 batches * 256 key tiles * 8 q-chunks = 8192 waves -> 1024 blocks
  k_denom<<<1024, 256, 0, stream>>>(Qb, Kb, denom);

  // 4 batches * 256 query tiles = 1024 waves -> 128 blocks
  k_out<<<128, 256, 0, stream>>>(Qb, Kb, Vt, denom, out);
}
